// CircuitGNN_53919019434527
// MI455X (gfx1250) — matrix operand order repaired and checked by reading, not compiled, not run on hardware
//
#include <hip/hip_runtime.h>
#include <hip/hip_bf16.h>

#define F_DIM   20
#define NN      100000
#define NE      3200000

typedef __attribute__((ext_vector_type(16))) __bf16 v16bf;
typedef __attribute__((ext_vector_type(8)))  float  v8f;

// A-fragment (16-bit, 16x32 MxK) K index for vector element e, given lane half.
// Per ISA 7.12.2: VGPR j (elements 2j,2j+1): K = 16*(j>>2) + 8*half + 2*(j&3) (+1)
__device__ __forceinline__ int a_k(int e, int half) {
    int j = e >> 1;
    return 16 * (j >> 2) + 8 * half + 2 * (j & 3) + (e & 1);
}

__device__ __forceinline__ v8f wmma_bf16(v16bf a, v16bf b, v8f c) {
    // (neg_a, A, neg_b, B, c_mod, C, reuse_a, reuse_b)
    return __builtin_amdgcn_wmma_f32_16x16x32_bf16(false, a, false, b, (short)0, c,
                                                   false, false);
}

// B-fragment (16-bit, 32x16 KxN): lane N = lane&15, same K order as A fragment,
// element e -> K = ktile*32 + a_k(e, half). Loads f32 weights W[in_dim][out_dim]
// row-major, converts to bf16, zero-pads out of range.
__device__ __forceinline__ v16bf load_bfrag(const float* __restrict__ W,
                                            int in_dim, int out_dim,
                                            int ktile, int ntile, int lane) {
    const int n    = ntile * 16 + (lane & 15);
    const int half = lane >> 4;
    v16bf b;
#pragma unroll
    for (int e = 0; e < 16; ++e) {
        const int k = ktile * 32 + a_k(e, half);
        float v = 0.0f;
        if (k < in_dim && n < out_dim) v = W[k * out_dim + n];
        b[e] = (__bf16)v;
    }
    return b;
}

__global__ void zero_kernel(float* __restrict__ p, long long n) {
    long long i = (long long)blockIdx.x * blockDim.x + threadIdx.x;
    if (i < n) p[i] = 0.0f;
}

// ---------------- Edge MLP + fused scatter-add aggregation -------------------
__global__ void __launch_bounds__(256)
edge_mlp_kernel(const float* __restrict__ nodes,
                const float* __restrict__ edge_feats,
                const int*   __restrict__ senders,
                const int*   __restrict__ receivers,
                const float* __restrict__ we0_w, const float* __restrict__ we0_b,
                const float* __restrict__ we1_w, const float* __restrict__ we1_b,
                const float* __restrict__ we2_w, const float* __restrict__ we2_b,
                float* __restrict__ agg)
{
    __shared__ __bf16 sh0[8][16 * 64];   // per-wave H0 tile, row-major [M][64]
    __shared__ __bf16 sh1[8][16 * 32];   // per-wave H1 tile, row-major [M][32]

    const int lane = threadIdx.x & 31;
    const int wave = threadIdx.x >> 5;
    const int half = lane >> 4;
    const int mrow = lane & 15;

    // Register-resident weight B-fragments, reused across the grid-stride loop.
    v16bf B0[2][4], B1[2][2], B2[2];
#pragma unroll
    for (int kt = 0; kt < 2; ++kt)
#pragma unroll
        for (int nt = 0; nt < 4; ++nt)
            B0[kt][nt] = load_bfrag(we0_w, 3 * F_DIM, 64, kt, nt, lane);
#pragma unroll
    for (int kt = 0; kt < 2; ++kt)
#pragma unroll
        for (int nt = 0; nt < 2; ++nt)
            B1[kt][nt] = load_bfrag(we1_w, 64, 32, kt, nt, lane);
#pragma unroll
    for (int nt = 0; nt < 2; ++nt)
        B2[nt] = load_bfrag(we2_w, 32, F_DIM, 0, nt, lane);

    const int n_tiles     = NE / 16;                 // 200000, exact
    const int wave_global = blockIdx.x * 8 + wave;
    const int n_waves     = gridDim.x * 8;

    for (int tile = wave_global; tile < n_tiles; tile += n_waves) {
        const long long ebase = (long long)tile * 16;
        const long long edge  = ebase + mrow;
        const int snd = senders[edge];
        const int rcv = receivers[edge];

        // Gather X = [edge_feats | nodes[snd] | nodes[rcv]] (16x60, pad to 64)
        // directly into A-fragment layout, cvt f32 -> bf16.
        v16bf a0, a1;
#pragma unroll
        for (int e = 0; e < 16; ++e) {
            const int k = a_k(e, half);              // 0..31
            float v;
            if (k < F_DIM)          v = edge_feats[edge * F_DIM + k];
            else                    v = nodes[(long long)snd * F_DIM + (k - F_DIM)];
            a0[e] = (__bf16)v;
        }
#pragma unroll
        for (int e = 0; e < 16; ++e) {
            const int k = 32 + a_k(e, half);         // 32..63
            float v = 0.0f;
            if (k < 2 * F_DIM)      v = nodes[(long long)snd * F_DIM + (k - F_DIM)];
            else if (k < 3 * F_DIM) v = nodes[(long long)rcv * F_DIM + (k - 2 * F_DIM)];
            a1[e] = (__bf16)v;
        }

        // Layer 0: [16x60] @ [60x64] + b, relu -> LDS
#pragma unroll
        for (int nt = 0; nt < 4; ++nt) {
            const float bias = we0_b[nt * 16 + mrow];
            v8f c = {bias, bias, bias, bias, bias, bias, bias, bias};
            c = wmma_bf16(a0, B0[0][nt], c);
            c = wmma_bf16(a1, B0[1][nt], c);
#pragma unroll
            for (int r = 0; r < 8; ++r) {
                const int m = r + 8 * half;          // C layout: M = r + 8*half
                sh0[wave][m * 64 + nt * 16 + mrow] = (__bf16)fmaxf(c[r], 0.0f);
            }
        }

        // Layer 1: [16x64] @ [64x32] + b, relu -> LDS
        v16bf h0a, h0b;
#pragma unroll
        for (int e = 0; e < 16; ++e) h0a[e] = sh0[wave][mrow * 64 + a_k(e, half)];
#pragma unroll
        for (int e = 0; e < 16; ++e) h0b[e] = sh0[wave][mrow * 64 + 32 + a_k(e, half)];
#pragma unroll
        for (int nt = 0; nt < 2; ++nt) {
            const float bias = we1_b[nt * 16 + mrow];
            v8f c = {bias, bias, bias, bias, bias, bias, bias, bias};
            c = wmma_bf16(h0a, B1[0][nt], c);
            c = wmma_bf16(h0b, B1[1][nt], c);
#pragma unroll
            for (int r = 0; r < 8; ++r) {
                const int m = r + 8 * half;
                sh1[wave][m * 32 + nt * 16 + mrow] = (__bf16)fmaxf(c[r], 0.0f);
            }
        }

        // Layer 2: [16x32] @ [32x20] + b -> messages, scatter-add into agg (L2-hot)
        v16bf h1a;
#pragma unroll
        for (int e = 0; e < 16; ++e) h1a[e] = sh1[wave][mrow * 32 + a_k(e, half)];
#pragma unroll
        for (int nt = 0; nt < 2; ++nt) {
            const int   n    = nt * 16 + mrow;
            const float bias = (n < F_DIM) ? we2_b[n] : 0.0f;
            v8f c = {bias, bias, bias, bias, bias, bias, bias, bias};
            c = wmma_bf16(h1a, B2[nt], c);
            if (n < F_DIM) {
#pragma unroll
                for (int r = 0; r < 8; ++r) {
                    const int m  = r + 8 * half;
                    const int rv = receivers[ebase + m];
                    unsafeAtomicAdd(&agg[(long long)rv * F_DIM + n], c[r]);
                }
            }
        }
    }
}

// ---------------------------- Node MLP update --------------------------------
__global__ void __launch_bounds__(256)
node_mlp_kernel(const float* __restrict__ nodes,
                const float* __restrict__ agg,
                const float* __restrict__ wn0_w, const float* __restrict__ wn0_b,
                const float* __restrict__ wn1_w, const float* __restrict__ wn1_b,
                const float* __restrict__ wn2_w, const float* __restrict__ wn2_b,
                float* __restrict__ out)
{
    __shared__ __bf16 sh0[8][16 * 64];
    __shared__ __bf16 sh1[8][16 * 32];

    const int lane = threadIdx.x & 31;
    const int wave = threadIdx.x >> 5;
    const int half = lane >> 4;
    const int mrow = lane & 15;

    v16bf B0[2][4], B1[2][2], B2[2];
#pragma unroll
    for (int kt = 0; kt < 2; ++kt)
#pragma unroll
        for (int nt = 0; nt < 4; ++nt)
            B0[kt][nt] = load_bfrag(wn0_w, 2 * F_DIM, 64, kt, nt, lane);
#pragma unroll
    for (int kt = 0; kt < 2; ++kt)
#pragma unroll
        for (int nt = 0; nt < 2; ++nt)
            B1[kt][nt] = load_bfrag(wn1_w, 64, 32, kt, nt, lane);
#pragma unroll
    for (int nt = 0; nt < 2; ++nt)
        B2[nt] = load_bfrag(wn2_w, 32, F_DIM, 0, nt, lane);

    const int n_tiles     = NN / 16;                 // 6250, exact
    const int wave_global = blockIdx.x * 8 + wave;
    const int n_waves     = gridDim.x * 8;

    for (int tile = wave_global; tile < n_tiles; tile += n_waves) {
        const long long nbase = (long long)tile * 16;
        const long long node  = nbase + mrow;

        // Y = [nodes | agg] (16x40, pad to 64) -> A fragments
        v16bf a0, a1;
#pragma unroll
        for (int e = 0; e < 16; ++e) {
            const int k = a_k(e, half);              // 0..31
            float v;
            if (k < F_DIM) v = nodes[node * F_DIM + k];
            else           v = agg[node * F_DIM + (k - F_DIM)];
            a0[e] = (__bf16)v;
        }
#pragma unroll
        for (int e = 0; e < 16; ++e) {
            const int k = 32 + a_k(e, half);         // 32..63
            float v = 0.0f;
            if (k < 2 * F_DIM) v = agg[node * F_DIM + (k - F_DIM)];
            a1[e] = (__bf16)v;
        }

#pragma unroll
        for (int nt = 0; nt < 4; ++nt) {
            const float bias = wn0_b[nt * 16 + mrow];
            v8f c = {bias, bias, bias, bias, bias, bias, bias, bias};
            c = wmma_bf16(a0, B0[0][nt], c);
            c = wmma_bf16(a1, B0[1][nt], c);
#pragma unroll
            for (int r = 0; r < 8; ++r) {
                const int m = r + 8 * half;
                sh0[wave][m * 64 + nt * 16 + mrow] = (__bf16)fmaxf(c[r], 0.0f);
            }
        }

        v16bf h0a, h0b;
#pragma unroll
        for (int e = 0; e < 16; ++e) h0a[e] = sh0[wave][mrow * 64 + a_k(e, half)];
#pragma unroll
        for (int e = 0; e < 16; ++e) h0b[e] = sh0[wave][mrow * 64 + 32 + a_k(e, half)];
#pragma unroll
        for (int nt = 0; nt < 2; ++nt) {
            const float bias = wn1_b[nt * 16 + mrow];
            v8f c = {bias, bias, bias, bias, bias, bias, bias, bias};
            c = wmma_bf16(h0a, B1[0][nt], c);
            c = wmma_bf16(h0b, B1[1][nt], c);
#pragma unroll
            for (int r = 0; r < 8; ++r) {
                const int m = r + 8 * half;
                sh1[wave][m * 32 + nt * 16 + mrow] = (__bf16)fmaxf(c[r], 0.0f);
            }
        }

        v16bf h1a;
#pragma unroll
        for (int e = 0; e < 16; ++e) h1a[e] = sh1[wave][mrow * 32 + a_k(e, half)];
#pragma unroll
        for (int nt = 0; nt < 2; ++nt) {
            const int n = nt * 16 + mrow;
            v8f c = {0.f, 0.f, 0.f, 0.f, 0.f, 0.f, 0.f, 0.f};
            c = wmma_bf16(h1a, B2[nt], c);
            if (n < F_DIM) {
                const float bias = wn2_b[n];
#pragma unroll
                for (int r = 0; r < 8; ++r) {
                    const int m = r + 8 * half;
                    out[(nbase + m) * F_DIM + n] = c[r] + bias;
                }
            }
        }
    }
}

extern "C" void kernel_launch(void* const* d_in, const int* in_sizes, int n_in,
                              void* d_out, int out_size, void* d_ws, size_t ws_size,
                              hipStream_t stream) {
    const float* nodes      = (const float*)d_in[0];
    const float* edge_feats = (const float*)d_in[1];
    const int*   senders    = (const int*)d_in[2];
    const int*   receivers  = (const int*)d_in[3];
    const float* we0_w = (const float*)d_in[4];
    const float* we0_b = (const float*)d_in[5];
    const float* we1_w = (const float*)d_in[6];
    const float* we1_b = (const float*)d_in[7];
    const float* we2_w = (const float*)d_in[8];
    const float* we2_b = (const float*)d_in[9];
    const float* wn0_w = (const float*)d_in[10];
    const float* wn0_b = (const float*)d_in[11];
    const float* wn1_w = (const float*)d_in[12];
    const float* wn1_b = (const float*)d_in[13];
    const float* wn2_w = (const float*)d_in[14];
    const float* wn2_b = (const float*)d_in[15];
    float* out = (float*)d_out;
    float* agg = (float*)d_ws;   // N_NODES * F floats = 8 MB scratch (L2-resident)

    const long long aggN = (long long)NN * F_DIM;
    zero_kernel<<<(int)((aggN + 255) / 256), 256, 0, stream>>>(agg, aggN);

    // 1024 blocks * 8 waves: each wave grid-strides ~25 sixteen-edge tiles,
    // amortizing the register-resident weight fragments.
    edge_mlp_kernel<<<1024, 256, 0, stream>>>(nodes, edge_feats, senders, receivers,
                                              we0_w, we0_b, we1_w, we1_b, we2_w, we2_b,
                                              agg);

    node_mlp_kernel<<<782, 256, 0, stream>>>(nodes, agg,
                                             wn0_w, wn0_b, wn1_w, wn1_b, wn2_w, wn2_b,
                                             out);
}
